// causal_kMemoryHMVAE_small_Dec_toeplitz_34703335752116
// MI455X (gfx1250) — matrix operand order repaired and checked by reading, not compiled, not run on hardware
//
#include <hip/hip_runtime.h>
#include <math.h>

typedef __attribute__((ext_vector_type(16))) _Float16 v16h;
typedef __attribute__((ext_vector_type(8)))  float    v8f;
typedef __attribute__((ext_vector_type(4)))  float    v4f;

#define BATCH 32768
#define BN_EPS 1e-5f

// ---------------------------------------------------------------------------
// Prep kernels
// ---------------------------------------------------------------------------
__global__ void pad_copy_z(const float* __restrict__ z, float* __restrict__ zp) {
  int i = blockIdx.x * blockDim.x + threadIdx.x;
  if (i >= BATCH * 32) return;
  int b = i >> 5, k = i & 31;
  zp[i] = (k < 24) ? z[b * 24 + k] : 0.f;
}

// W is (K x N) row-major f32; Wt is (Npad x Kpad) row-major f16 (transposed, zero-padded)
__global__ void convert_weight(const float* __restrict__ W, _Float16* __restrict__ Wt,
                               int K, int N, int Kpad, int Npad) {
  int i = blockIdx.x * blockDim.x + threadIdx.x;
  if (i >= Npad * Kpad) return;
  int n = i / Kpad, k = i % Kpad;
  float v = (k < K && n < N) ? W[k * N + n] : 0.f;
  Wt[i] = (_Float16)v;
}

// ---------------------------------------------------------------------------
// Deterministic per-column batch stats -> affine BN params (scale, shift)
// scale = g / sqrt(var+eps); shift = beta - mean*scale
// ---------------------------------------------------------------------------
__global__ __launch_bounds__(256) void col_stats(
    const float* __restrict__ Y, int ld, int Nact,
    const float* __restrict__ gamma, const float* __restrict__ beta,
    float* __restrict__ scale, float* __restrict__ shift) {
  __shared__ float s1[256], s2[256];
  int c = blockIdx.x;
  int t = threadIdx.x;
  float a = 0.f, b = 0.f;
  for (int r = t; r < BATCH; r += 256) {
    float v = Y[(size_t)r * ld + c];
    a += v; b += v * v;
  }
  s1[t] = a; s2[t] = b;
  __syncthreads();
  for (int s = 128; s > 0; s >>= 1) {
    if (t < s) { s1[t] += s1[t + s]; s2[t] += s2[t + s]; }
    __syncthreads();
  }
  if (t == 0) {
    float mean = s1[0] * (1.0f / BATCH);
    float var  = s2[0] * (1.0f / BATCH) - mean * mean;   // biased var, like jnp.var
    float g  = (c < Nact) ? gamma[c] : 0.f;
    float be = (c < Nact) ? beta[c]  : 0.f;
    float sc = g * rsqrtf(var + BN_EPS);
    scale[c] = sc;
    shift[c] = be - mean * sc;
  }
}

// ---------------------------------------------------------------------------
// WMMA GEMM: Y(B x Npad) = act( [BN(X)](B x Kpad) * W^T(Npad x Kpad) + bias )
// Block: 256 threads = 8 waves; tile 64 rows x 32 cols; each wave: one 16x16 C.
// A fragment (16x32 f16, ISA layout): lane l: M=l%16, g=l/16,
//   e=0..7  -> K = kt + g*8 + e;  e=8..15 -> K = kt + 16 + g*8 + (e-8)
// B fragment (32x16 f16): lane l: N=l%16, same K mapping as A
// C/D (16x16 f32): vgpr v, lane l: M = v + 8*(l/16), N = l%16
// ---------------------------------------------------------------------------
template <bool RELU, bool BN>
__global__ __launch_bounds__(256) void gemm_wmma(
    const float* __restrict__ X, int ldx,
    const _Float16* __restrict__ Wt,
    const float* __restrict__ bias, int Nact,
    const float* __restrict__ scale, const float* __restrict__ shift,
    float* __restrict__ Y, int ldy, int Kpad) {
  const int lane = threadIdx.x & 31;
  const int wave = threadIdx.x >> 5;
  const int g    = lane >> 4;
  const int mn   = lane & 15;
  const int row0 = blockIdx.y * 64 + (wave >> 1) * 16;
  const int col0 = blockIdx.x * 32 + (wave & 1) * 16;

  v8f acc = {};
  const float*    xrow = X  + (size_t)(row0 + mn) * ldx;
  const _Float16* wrow = Wt + (size_t)(col0 + mn) * Kpad;

  for (int kt = 0; kt < Kpad; kt += 32) {
    v4f x0 = *(const v4f*)(xrow + kt + g * 8);
    v4f x1 = *(const v4f*)(xrow + kt + g * 8 + 4);
    v4f x2 = *(const v4f*)(xrow + kt + 16 + g * 8);
    v4f x3 = *(const v4f*)(xrow + kt + 16 + g * 8 + 4);
    if (BN) {
      const float* sp = scale + kt + g * 8;
      const float* tp = shift + kt + g * 8;
      v4f s0 = *(const v4f*)(sp);      v4f t0 = *(const v4f*)(tp);
      v4f s1 = *(const v4f*)(sp + 4);  v4f t1 = *(const v4f*)(tp + 4);
      v4f s2 = *(const v4f*)(sp + 16); v4f t2 = *(const v4f*)(tp + 16);
      v4f s3 = *(const v4f*)(sp + 20); v4f t3 = *(const v4f*)(tp + 20);
      x0 = x0 * s0 + t0; x1 = x1 * s1 + t1;
      x2 = x2 * s2 + t2; x3 = x3 * s3 + t3;
    }
    v16h afrag;
#pragma unroll
    for (int e = 0; e < 4; ++e) {
      afrag[e]      = (_Float16)x0[e];
      afrag[e + 4]  = (_Float16)x1[e];
      afrag[e + 8]  = (_Float16)x2[e];
      afrag[e + 12] = (_Float16)x3[e];
    }
    typedef __attribute__((ext_vector_type(8))) _Float16 v8h_;
    v8h_ blo = *(const v8h_*)(wrow + kt + g * 8);
    v8h_ bhi = *(const v8h_*)(wrow + kt + 16 + g * 8);
    v16h bfrag = __builtin_shufflevector(blo, bhi, 0, 1, 2, 3, 4, 5, 6, 7, 8, 9, 10, 11, 12, 13, 14, 15);
    acc = __builtin_amdgcn_wmma_f32_16x16x32_f16(
        /*neg_a=*/false, afrag, /*neg_b=*/false, bfrag,
        /*c_mod=*/(short)0, acc, /*reuse_a=*/false, /*reuse_b=*/false);
  }

  const int colN = col0 + mn;
  const float bv = (colN < Nact) ? bias[colN] : 0.f;
#pragma unroll
  for (int v = 0; v < 8; ++v) {
    float yv = acc[v] + bv;
    if (RELU) yv = fmaxf(yv, 0.f);
    Y[(size_t)(row0 + v + 8 * g) * ldy + colN] = yv;
  }
}

// ---------------------------------------------------------------------------
// Alpha + Toeplitz expansion. One wave per batch row.
//   a0 = min(exp(a[0]),500); Alpha[k] = 0.022*a0*(tanh(a[k]) + i*tanh(a[31+k])), k=1..31
//   Bmat[i][j] = Alpha[i-j]            (i>=j, else 0)
//   Cmat[i][j] = conj(Alpha[32-(i-j)]) (i> j, else 0)
// ---------------------------------------------------------------------------
__global__ __launch_bounds__(256) void alpha_toeplitz(
    const float* __restrict__ A, float* __restrict__ outB, float* __restrict__ outC) {
  __shared__ float sRe[8][32];
  __shared__ float sIm[8][32];
  const int w    = threadIdx.x >> 5;
  const int lane = threadIdx.x & 31;
  const int b    = blockIdx.x * 8 + w;
  const float* arow = A + (size_t)b * 64;

  float a0 = fminf(expf(arow[0]), 500.f);
  float re, im;
  if (lane == 0) { re = a0; im = 0.f; }
  else {
    re = 0.022f * a0 * tanhf(arow[lane]);
    im = 0.022f * a0 * tanhf(arow[31 + lane]);
  }
  sRe[w][lane] = re; sIm[w][lane] = im;
  __syncthreads();

  float2* Bp = (float2*)(outB + (size_t)b * 2048);
  float2* Cp = (float2*)(outC + (size_t)b * 2048);
  const int j = lane;
#pragma unroll 4
  for (int i = 0; i < 32; ++i) {
    int d = i - j;
    float2 bv = make_float2(0.f, 0.f), cv = make_float2(0.f, 0.f);
    if (d >= 0) { bv.x = sRe[w][d];      bv.y = sIm[w][d]; }
    if (d >= 1) { cv.x = sRe[w][32 - d]; cv.y = -sIm[w][32 - d]; }
    Bp[i * 32 + j] = bv;
    Cp[i * 32 + j] = cv;
  }
}

// ---------------------------------------------------------------------------
// Launcher
// ---------------------------------------------------------------------------
extern "C" void kernel_launch(void* const* d_in, const int* in_sizes, int n_in,
                              void* d_out, int out_size, void* d_ws, size_t ws_size,
                              hipStream_t stream) {
  (void)in_sizes; (void)n_in; (void)out_size; (void)ws_size;

  const float* z   = (const float*)d_in[0];
  const float* W1  = (const float*)d_in[1];
  const float* b1  = (const float*)d_in[2];
  const float* g1  = (const float*)d_in[3];
  const float* be1 = (const float*)d_in[4];
  const float* W2  = (const float*)d_in[5];
  const float* b2  = (const float*)d_in[6];
  const float* g2  = (const float*)d_in[7];
  const float* be2 = (const float*)d_in[8];
  const float* W3  = (const float*)d_in[9];
  const float* b3  = (const float*)d_in[10];
  const float* Wa1 = (const float*)d_in[11];
  const float* ba1 = (const float*)d_in[12];
  const float* ga  = (const float*)d_in[13];
  const float* bea = (const float*)d_in[14];
  const float* Wa2 = (const float*)d_in[15];
  const float* ba2 = (const float*)d_in[16];

  char* ws = (char*)d_ws;
  size_t off = 0;
  auto alloc = [&](size_t bytes) -> void* {
    void* p = ws + off;
    off = (off + bytes + 255) & ~(size_t)255;
    return p;
  };

  float* zp   = (float*)alloc((size_t)BATCH * 32  * 4);
  float* pre1 = (float*)alloc((size_t)BATCH * 320 * 4);
  float* pre2 = (float*)alloc((size_t)BATCH * 416 * 4);
  float* prea = (float*)alloc((size_t)BATCH * 160 * 4);
  float* aout = (float*)alloc((size_t)BATCH * 64  * 4);
  _Float16* W1h  = (_Float16*)alloc((size_t)320  * 32  * 2);
  _Float16* W2h  = (_Float16*)alloc((size_t)416  * 320 * 2);
  _Float16* W3h  = (_Float16*)alloc((size_t)1024 * 416 * 2);
  _Float16* Wa1h = (_Float16*)alloc((size_t)160  * 320 * 2);
  _Float16* Wa2h = (_Float16*)alloc((size_t)64   * 160 * 2);
  float* sc1 = (float*)alloc(320 * 4); float* sh1 = (float*)alloc(320 * 4);
  float* sc2 = (float*)alloc(416 * 4); float* sh2 = (float*)alloc(416 * 4);
  float* scA = (float*)alloc(160 * 4); float* shA = (float*)alloc(160 * 4);

  float* outMu = (float*)d_out;
  float* outB  = outMu + (size_t)BATCH * 1024;
  float* outC  = outB  + (size_t)BATCH * 2048;

  // ---- prep: pad z, convert + transpose + pad weights to f16 ----
  pad_copy_z<<<(BATCH * 32 + 255) / 256, 256, 0, stream>>>(z, zp);
  convert_weight<<<(320 * 32   + 255) / 256, 256, 0, stream>>>(W1,  W1h,  24,  300, 32,  320);
  convert_weight<<<(416 * 320  + 255) / 256, 256, 0, stream>>>(W2,  W2h,  300, 400, 320, 416);
  convert_weight<<<(1024 * 416 + 255) / 256, 256, 0, stream>>>(W3,  W3h,  400, 1024, 416, 1024);
  convert_weight<<<(160 * 320  + 255) / 256, 256, 0, stream>>>(Wa1, Wa1h, 300, 150, 320, 160);
  convert_weight<<<(64 * 160   + 255) / 256, 256, 0, stream>>>(Wa2, Wa2h, 150, 63,  160, 64);

  const dim3 blk(256);
  const int gy = BATCH / 64;  // 512

  // ---- layer 1: h_pre = relu(z @ W1 + b1), then BN1 params ----
  gemm_wmma<true, false><<<dim3(10, gy), blk, 0, stream>>>(
      zp, 32, W1h, b1, 300, nullptr, nullptr, pre1, 320, 32);
  col_stats<<<320, 256, 0, stream>>>(pre1, 320, 300, g1, be1, sc1, sh1);

  // ---- layer 2: m_pre = relu(BN1(h) @ W2 + b2), then BN2 params ----
  gemm_wmma<true, true><<<dim3(13, gy), blk, 0, stream>>>(
      pre1, 320, W2h, b2, 400, sc1, sh1, pre2, 416, 320);
  col_stats<<<416, 256, 0, stream>>>(pre2, 416, 400, g2, be2, sc2, sh2);

  // ---- mu_out = BN2(m) @ W3 + b3  (written directly to d_out) ----
  gemm_wmma<false, true><<<dim3(32, gy), blk, 0, stream>>>(
      pre2, 416, W3h, b3, 1024, sc2, sh2, outMu, 1024, 416);

  // ---- a-branch: relu(BN1(h) @ Wa1 + ba1), BNa, @ Wa2 + ba2 ----
  gemm_wmma<true, true><<<dim3(5, gy), blk, 0, stream>>>(
      pre1, 320, Wa1h, ba1, 150, sc1, sh1, prea, 160, 320);
  col_stats<<<160, 256, 0, stream>>>(prea, 160, 150, ga, bea, scA, shA);
  gemm_wmma<false, true><<<dim3(2, gy), blk, 0, stream>>>(
      prea, 160, Wa2h, ba2, 63, scA, shA, aout, 64, 160);

  // ---- Alpha + Toeplitz expansion (dominant memory writer) ----
  alpha_toeplitz<<<BATCH / 8, 256, 0, stream>>>(aout, outB, outC);
}
